// CapsNet4Sequence_84224308675201
// MI455X (gfx1250) — matrix operand order repaired and checked by reading, not compiled, not run on hardware
//
#include <hip/hip_runtime.h>
#include <hip/hip_bf16.h>

// ---------------------------------------------------------------------------
// CDNA5 (gfx1250): bf16 WMMA (V_WMMA_F32_16X16X32_BF16) for all GEMMs with
// fp32 accumulation; TDM (tensor_load_to_lds) double-buffered gate prefetch
// for the sequential LSTM recurrence.
// ---------------------------------------------------------------------------

typedef __attribute__((ext_vector_type(16))) __bf16 v16bf;
typedef __attribute__((ext_vector_type(8)))  float  v8f;
typedef unsigned int u32x4 __attribute__((ext_vector_type(4)));
typedef int          i32x8 __attribute__((ext_vector_type(8)));
typedef int          i32x4 __attribute__((ext_vector_type(4)));

#define H2C 256   // per-direction hidden

__device__ __forceinline__ unsigned short f2bf(float f) {
  union { float f; unsigned u; } v; v.f = f;
  unsigned r = v.u + 0x7FFFu + ((v.u >> 16) & 1u);   // round-to-nearest-even
  return (unsigned short)(r >> 16);
}

union V16U { v16bf v; uint4 q[2]; };

__device__ __forceinline__ v8f v8f_zero() {
  v8f z = {0.f, 0.f, 0.f, 0.f, 0.f, 0.f, 0.f, 0.f};
  return z;
}

// A tile 16x32 bf16 (MxK). lanes 0-15: M=lane, VGPR0-3 = K 0..7, VGPR4-7 = K 16..23.
// lanes 16-31: M=lane-16, VGPR0-3 = K 8..15, VGPR4-7 = K 24..31.  (ISA 7.12.2)
__device__ __forceinline__ v16bf load_a16x32(const unsigned short* Arow0, int lda,
                                             int k0, int lane) {
  V16U u;
  const unsigned short* p =
      Arow0 + (size_t)(lane & 15) * lda + k0 + ((lane & 16) ? 8 : 0);
  u.q[0] = *(const uint4*)(p);
  u.q[1] = *(const uint4*)(p + 16);
  return u.v;
}

// B tile 32x16 (KxN) from a weight stored row-major [N, K] (computes A.B^T).
// lane holds column n = lane&15; lanes 0-15: K 0..7,16..23 ; lanes 16-31: K 8..15,24..31.
__device__ __forceinline__ v16bf load_b32x16(const unsigned short* Brow0, int ldb,
                                             int k0, int lane) {
  V16U u;
  const unsigned short* p =
      Brow0 + (size_t)(lane & 15) * ldb + k0 + ((lane & 16) ? 8 : 0);
  u.q[0] = *(const uint4*)(p);
  u.q[1] = *(const uint4*)(p + 16);
  return u.v;
}

__device__ __forceinline__ v8f wmma_bf16(v16bf a, v16bf b, v8f c) {
  return __builtin_amdgcn_wmma_f32_16x16x32_bf16(false, a, false, b,
                                                 (short)0, c, false, false);
}

// ---------------------------------------------------------------------------
// Tensor Data Mover: 2D tile load Global -> LDS (D# per ISA 8.3/8.4).
// dim0 = contiguous elements per row; dim1 = rows; row stride in elements.
// data_size code: 0=1B,1=2B,2=4B,3=8B.
// clang-23 toolchain: 6-arg builtin (v4u, v8i, v4i, v4i, v8i, i32 cpol).
// ---------------------------------------------------------------------------
__device__ __forceinline__ void tdm_load_2d(unsigned lds_off, const void* gptr,
                                            unsigned tile_d0, unsigned tile_d1,
                                            unsigned long row_stride_elems,
                                            unsigned data_size_code) {
  unsigned long ga = (unsigned long)gptr;
  u32x4 g0;
  g0[0] = 1u;                                        // count=1, user D#
  g0[1] = lds_off;                                   // lds_addr (bytes)
  g0[2] = (unsigned)(ga & 0xFFFFFFFFu);              // global_addr[31:0]
  g0[3] = (unsigned)((ga >> 32) & 0x01FFFFFFu)       // global_addr[56:32]
        | 0x80000000u;                               // type=2 ("image")
  i32x8 g1;
  g1[0] = (int)(data_size_code << 16);               // wg_mask=0 | data_size
  g1[1] = (int)((tile_d0 & 0xFFFFu) << 16);          // tensor_dim0[15:0]
  g1[2] = (int)(((tile_d0 >> 16) & 0xFFFFu)          // tensor_dim0[31:16]
        | ((tile_d1 & 0xFFFFu) << 16));              // tensor_dim1[15:0]
  g1[3] = (int)(((tile_d1 >> 16) & 0xFFFFu)          // tensor_dim1[31:16]
        | ((tile_d0 & 0xFFFFu) << 16));              // tile_dim0
  g1[4] = (int)(tile_d1 & 0xFFFFu);                  // tile_dim1 (tile_dim2=0)
  g1[5] = (int)(row_stride_elems & 0xFFFFFFFFu);     // tensor_dim0_stride[31:0]
  g1[6] = (int)((row_stride_elems >> 32) & 0xFFFFu); // stride[47:32]
  g1[7] = 0;
  i32x4 z4 = {0, 0, 0, 0};                           // groups 2/3 unused (2D)
  i32x8 z8 = {0, 0, 0, 0, 0, 0, 0, 0};
  __builtin_amdgcn_tensor_load_to_lds(g0, g1, z4, z4, z8, 0);
}

// ---------------------------------------------------------------------------
// fp32 -> bf16 pack with optional K padding (Kin -> Kpad, zeros in the pad)
// ---------------------------------------------------------------------------
__global__ void pad_cvt(const float* __restrict__ src, unsigned short* __restrict__ dst,
                        int rows, int Kin, int Kpad) {
  long i = (long)blockIdx.x * blockDim.x + threadIdx.x;
  if (i >= (long)rows * Kpad) return;
  int r = (int)(i / Kpad), k = (int)(i % Kpad);
  float v = (k < Kin) ? src[(long)r * Kin + k] : 0.f;
  dst[i] = f2bf(v);
}

// ---------------------------------------------------------------------------
// Embedding gather -> bf16, E=300 padded to 320
// ---------------------------------------------------------------------------
__global__ void embed_gather(const int* __restrict__ tok, const float* __restrict__ emb,
                             unsigned short* __restrict__ e_bf, long total) {
  long i = (long)blockIdx.x * blockDim.x + threadIdx.x;
  if (i >= total) return;
  long row = i / 320;
  int  k   = (int)(i % 320);
  int  t   = tok[row];
  float v  = (k < 300) ? emb[(long)t * 300 + k] : 0.f;
  e_bf[i] = f2bf(v);
}

// ---------------------------------------------------------------------------
// Generic batched bf16 WMMA GEMM (1 tile/wave): C[z][m,n] = sum_k A[m,k]*B[n,k]
// Used for the batched capsule projections (N small).
// ---------------------------------------------------------------------------
__global__ void __launch_bounds__(128)
gemm_bf16(const unsigned short* __restrict__ A, const unsigned short* __restrict__ B,
          float* __restrict__ C, int K, int ldc, int Nvalid,
          long sA, long sB, long sC) {
  const int lane = threadIdx.x & 31;
  const int wave = threadIdx.x >> 5;
  const int m0 = blockIdx.x * 16;
  const int n0 = (blockIdx.y * 4 + wave) * 16;
  const unsigned short* Ab = A + (long)blockIdx.z * sA + (size_t)m0 * K;
  const unsigned short* Bb = B + (long)blockIdx.z * sB + (size_t)n0 * K;
  v8f acc = v8f_zero();
  for (int k0 = 0; k0 < K; k0 += 32) {
    __builtin_prefetch(Ab + (size_t)(lane & 15) * K + k0 + 64, 0, 1);
    v16bf a = load_a16x32(Ab, K, k0, lane);
    v16bf b = load_b32x16(Bb, K, k0, lane);
    acc = wmma_bf16(a, b, acc);
  }
  const int col = n0 + (lane & 15);
  if (col < Nvalid) {
    float* Cb = C + (long)blockIdx.z * sC;
    const int mb = m0 + ((lane & 16) ? 8 : 0);
#pragma unroll
    for (int r = 0; r < 8; ++r)
      Cb[(long)(mb + r) * ldc + col] = acc[r];
  }
}

// ---------------------------------------------------------------------------
// Register-blocked GEMM: each wave owns 16x64 (4 accumulator tiles) so every
// A fragment is reused 4x in-register (4 WMMA per A load). Used for the big
// input-projection GEMMs (N = 2048, multiple of 256 per block).
// grid: (M/16, N/256), block 128.
// ---------------------------------------------------------------------------
__global__ void __launch_bounds__(128)
gemm_bf16_x4(const unsigned short* __restrict__ A, const unsigned short* __restrict__ B,
             float* __restrict__ C, int K, int ldc) {
  const int lane = threadIdx.x & 31;
  const int wave = threadIdx.x >> 5;
  const int m0 = blockIdx.x * 16;
  const int n0 = (blockIdx.y * 4 + wave) * 64;
  const unsigned short* Ab = A + (size_t)m0 * K;
  v8f acc0 = v8f_zero(), acc1 = v8f_zero(), acc2 = v8f_zero(), acc3 = v8f_zero();
  for (int k0 = 0; k0 < K; k0 += 32) {
    __builtin_prefetch(Ab + (size_t)(lane & 15) * K + k0 + 64, 0, 1);
    v16bf a  = load_a16x32(Ab, K, k0, lane);
    v16bf b0 = load_b32x16(B + (size_t)(n0 +  0) * K, K, k0, lane);
    v16bf b1 = load_b32x16(B + (size_t)(n0 + 16) * K, K, k0, lane);
    v16bf b2 = load_b32x16(B + (size_t)(n0 + 32) * K, K, k0, lane);
    v16bf b3 = load_b32x16(B + (size_t)(n0 + 48) * K, K, k0, lane);
    acc0 = wmma_bf16(a, b0, acc0);
    acc1 = wmma_bf16(a, b1, acc1);
    acc2 = wmma_bf16(a, b2, acc2);
    acc3 = wmma_bf16(a, b3, acc3);
  }
  const int colb = lane & 15;
  const int mb = m0 + ((lane & 16) ? 8 : 0);
  v8f accs[4] = {acc0, acc1, acc2, acc3};
#pragma unroll
  for (int q = 0; q < 4; ++q)
#pragma unroll
    for (int r = 0; r < 8; ++r)
      C[(long)(mb + r) * ldc + n0 + q * 16 + colb] = accs[q][r];
}

// ---------------------------------------------------------------------------
// Bidirectional LSTM recurrence (one direction per blockIdx.y).
// xp: precomputed input projections [Nrows*L, 2048] fp32 (fwd 0..1023, bwd 1024..2047)
// Whh: bf16 [2048, 256]; gate order i,f,g,o (256 each).
// h_out: bf16 [Nrows, 64, 512] (fwd cols 0..255, bwd 256..511).
// Block = 16 batch rows, 8 waves; wave w owns hidden units [32w,32w+32) x 4 gates
// so gate math is lane-local on the WMMA accumulators. The 64KB xp gate slab
// for step t+1 is TDM-DMA'd into LDS (double buffered) while step t's WMMA runs.
// ---------------------------------------------------------------------------
__global__ void __launch_bounds__(256)
lstm_wmma(const float* __restrict__ xp, const unsigned short* __restrict__ Whh,
          const float* __restrict__ bias_f, const float* __restrict__ bias_b,
          unsigned short* __restrict__ h_out, int Nrows, int L) {
  __shared__ __align__(16) float          xbuf[2][16 * 1024];   // 128 KB (TDM dest)
  __shared__ __align__(16) unsigned short h_cur[16][H2C];       // 8 KB (WMMA A)
  const int lane = threadIdx.x & 31;
  const int wave = threadIdx.x >> 5;
  const int dir  = blockIdx.y;
  const int n0   = blockIdx.x * 16;
  const float* bias = dir ? bias_b : bias_f;
  const unsigned short* WhhD = Whh + (size_t)dir * 1024 * H2C;

  for (int i = threadIdx.x; i < 16 * H2C; i += 256) (&h_cur[0][0])[i] = 0;

  const int j0 = wave * 32;
  const int jc = lane & 15;
  const int rbase = (lane & 16) ? 8 : 0;

  float bv[2][4];
#pragma unroll
  for (int jt = 0; jt < 2; ++jt) {
    int j = j0 + jt * 16 + jc;
#pragma unroll
    for (int g = 0; g < 4; ++g) bv[jt][g] = bias[g * H2C + j];
  }
  float cst[2][8];
#pragma unroll
  for (int jt = 0; jt < 2; ++jt)
#pragma unroll
    for (int r = 0; r < 8; ++r) cst[jt][r] = 0.f;

  // Preload step-0 gate slab via TDM (wave 0 issues; TENSORcnt tracks it).
  if (wave == 0) {
    const int trd0 = dir ? (L - 1) : 0;
    tdm_load_2d((unsigned)(uintptr_t)(&xbuf[0][0]),
                xp + ((long)n0 * L + trd0) * 2048 + (long)dir * 1024,
                1024u, 16u, (unsigned long)L * 2048u, 2u);
    __builtin_amdgcn_s_wait_tensorcnt(0);
  }
  __syncthreads();

  for (int t = 0; t < L; ++t) {
    const int trd = dir ? (L - 1 - t) : t;
    // Kick off DMA for step t+1 into the other buffer; overlaps the WMMAs below.
    if (wave == 0 && t + 1 < L) {
      const int trdn = dir ? (L - 2 - t) : (t + 1);
      tdm_load_2d((unsigned)(uintptr_t)(&xbuf[(t + 1) & 1][0]),
                  xp + ((long)n0 * L + trdn) * 2048 + (long)dir * 1024,
                  1024u, 16u, (unsigned long)L * 2048u, 2u);
    }

    v8f acc[2][4];
#pragma unroll
    for (int jt = 0; jt < 2; ++jt)
#pragma unroll
      for (int g = 0; g < 4; ++g) acc[jt][g] = v8f_zero();

    for (int k0 = 0; k0 < H2C; k0 += 32) {
      v16bf a = load_a16x32(&h_cur[0][0], H2C, k0, lane);
#pragma unroll
      for (int jt = 0; jt < 2; ++jt)
#pragma unroll
        for (int g = 0; g < 4; ++g) {
          v16bf b = load_b32x16(WhhD + (size_t)(g * H2C + j0 + jt * 16) * H2C,
                                H2C, k0, lane);
          acc[jt][g] = wmma_bf16(a, b, acc[jt][g]);
        }
    }
    __syncthreads();   // all WMMA reads of h_cur done before overwrite

    const float* xs = &xbuf[t & 1][0];
#pragma unroll
    for (int jt = 0; jt < 2; ++jt) {
      const int j = j0 + jt * 16 + jc;
#pragma unroll
      for (int r = 0; r < 8; ++r) {
        const int  m = rbase + r;
        const long n = n0 + m;
        const float* xr = xs + m * 1024;
        float gi = acc[jt][0][r] + xr[0 * H2C + j] + bv[jt][0];
        float gf = acc[jt][1][r] + xr[1 * H2C + j] + bv[jt][1];
        float gg = acc[jt][2][r] + xr[2 * H2C + j] + bv[jt][2];
        float go = acc[jt][3][r] + xr[3 * H2C + j] + bv[jt][3];
        float si = 1.f / (1.f + __expf(-gi));   // trans ops co-execute with XDL WMMA
        float sf = 1.f / (1.f + __expf(-gf));
        float so = 1.f / (1.f + __expf(-go));
        float c  = sf * cst[jt][r] + si * tanhf(gg);
        float h  = so * tanhf(c);
        cst[jt][r] = c;
        unsigned short hb = f2bf(h);
        h_cur[m][j] = hb;
        h_out[((long)n * 64 + trd) * 512 + (size_t)dir * 256 + j] = hb;
      }
    }
    if (wave == 0) __builtin_amdgcn_s_wait_tensorcnt(0);  // next slab resident
    __syncthreads();
  }
}

// ---------------------------------------------------------------------------
// Dynamic-routing capsule layer. One block per sequence.
// wu: [batch, 256*L] fp32 == Wu[o, l] row-major. The reference's "faithful"
// no-permute reshape is a pure flat re-read: u_hat[d,l,f] = wu[l*256 + d*16 + f].
// ---------------------------------------------------------------------------
__global__ void __launch_bounds__(256)
caps_route(const float* __restrict__ wu, int L,
           float* __restrict__ cap_f, unsigned short* __restrict__ cap_b) {
  __shared__ float cb[16 * 64];
  __shared__ float bb[16 * 64];
  __shared__ float sS[256];
  __shared__ float sO[256];
  const int tid = threadIdx.x;
  const int d = tid >> 4, f = tid & 15;
  const float* U = wu + (long)blockIdx.x * 256 * L;

  for (int i = tid; i < 16 * 64; i += 256) bb[i] = 0.f;
  __syncthreads();

  float out_df = 0.f;
  for (int it = 0; it < 3; ++it) {
    if (tid < L) {                         // softmax over capsule dim (axis=1) per l
      float mx = -1e30f;
      for (int dd = 0; dd < 16; ++dd) mx = fmaxf(mx, bb[dd * 64 + tid]);
      float e[16], s = 0.f;
      for (int dd = 0; dd < 16; ++dd) { e[dd] = __expf(bb[dd * 64 + tid] - mx); s += e[dd]; }
      float inv = 1.f / s;
      for (int dd = 0; dd < 16; ++dd) cb[dd * 64 + tid] = e[dd] * inv;
    }
    __syncthreads();
    float sv = 0.f;                        // s[d][f] = sum_l u_hat[d,l,f] * c[d,l]
    for (int l = 0; l < L; ++l)
      sv += U[l * 256 + tid] * cb[d * 64 + l];
    sS[tid] = sv;
    __syncthreads();
    float ssq = 0.f;                       // squash over axis=1 (capsule dim d)
    for (int dd = 0; dd < 16; ++dd) { float x = sS[dd * 16 + f]; ssq += x * x; }
    float scale = (ssq / (1.f + ssq)) * (1.f / sqrtf(ssq));
    out_df = sv * scale;
    sO[tid] = out_df;
    __syncthreads();
    if (it < 2) {
      if (tid < L) {                       // b[d][l] += sum_f u_hat[d,l,f] * out[d,f]
        for (int dd = 0; dd < 16; ++dd) {
          float acc = 0.f;
          for (int ff = 0; ff < 16; ++ff)
            acc += U[tid * 256 + dd * 16 + ff] * sO[dd * 16 + ff];
          bb[dd * 64 + tid] += acc;
        }
      }
      __syncthreads();
    }
  }
  cap_f[(long)blockIdx.x * 256 + tid] = out_df;
  cap_b[(long)blockIdx.x * 256 + tid] = f2bf(out_df);
}

// ---------------------------------------------------------------------------
// Final FC: [128, 256] x [5, 256]^T + b -> [128, 5]
// ---------------------------------------------------------------------------
__global__ void fc_kernel(const float* __restrict__ cap, const float* __restrict__ W,
                          const float* __restrict__ b, float* __restrict__ out) {
  int idx = blockIdx.x * blockDim.x + threadIdx.x;
  if (idx >= 128 * 5) return;
  int bi = idx / 5, c = idx % 5;
  float a = b[c];
  for (int j = 0; j < 256; ++j) a += cap[bi * 256 + j] * W[c * 256 + j];
  out[idx] = a;
}

// ---------------------------------------------------------------------------
extern "C" void kernel_launch(void* const* d_in, const int* in_sizes, int n_in,
                              void* d_out, int out_size, void* d_ws, size_t ws_size,
                              hipStream_t stream) {
  (void)in_sizes; (void)n_in; (void)out_size; (void)ws_size;

  const int*   tok    = (const int*)  d_in[0];
  const float* embedW = (const float*)d_in[1];
  const float* Wih_f0 = (const float*)d_in[2];
  const float* Whh_f0 = (const float*)d_in[3];
  const float* b_f0   = (const float*)d_in[4];
  const float* Wih_b0 = (const float*)d_in[5];
  const float* Whh_b0 = (const float*)d_in[6];
  const float* b_b0   = (const float*)d_in[7];
  const float* Wih_f1 = (const float*)d_in[8];
  const float* Whh_f1 = (const float*)d_in[9];
  const float* b_f1   = (const float*)d_in[10];
  const float* Wih_b1 = (const float*)d_in[11];
  const float* Whh_b1 = (const float*)d_in[12];
  const float* b_b1   = (const float*)d_in[13];
  const float* W_caps = (const float*)d_in[14];
  const float* fcW    = (const float*)d_in[15];
  const float* fcb    = (const float*)d_in[16];
  float* out = (float*)d_out;

  unsigned char* ws = (unsigned char*)d_ws;
  size_t off = 0;
  auto take = [&](size_t bytes) -> void* {
    void* p = ws + off;
    off = (off + bytes + 255) & ~(size_t)255;
    return p;
  };

  unsigned short* WihFB0 = (unsigned short*)take(2048L * 320 * 2);
  unsigned short* WhhFB0 = (unsigned short*)take(2048L * 256 * 2);
  unsigned short* WihFB1 = (unsigned short*)take(2048L * 256 * 2);
  unsigned short* WhhFB1 = (unsigned short*)take(2048L * 256 * 2);
  unsigned short* WcapsB = (unsigned short*)take(256L * 512 * 2);
  unsigned short* E_BF   = (unsigned short*)take(153600L * 320 * 2);
  float*          XP0    = (float*)take(153600L * 2048 * 4);
  unsigned short* H1_BF  = (unsigned short*)take(2560L * 64 * 512 * 2);
  float*          WU1    = (float*)take(2560L * 256 * 60 * 4);
  float*          CAP1F  = (float*)take(2560L * 256 * 4);
  unsigned short* CAP1B  = (unsigned short*)take(2560L * 256 * 2);
  float*          XP1    = (float*)take(2560L * 2048 * 4);
  unsigned short* H2_BF  = (unsigned short*)take(128L * 64 * 512 * 2);
  float*          WU2    = (float*)take(128L * 256 * 20 * 4);
  float*          CAP2F  = (float*)take(128L * 256 * 4);
  unsigned short* CAP2B  = (unsigned short*)take(128L * 256 * 2);

  // --- weight packing (fp32 -> bf16, pad E 300->320) ---
  #define CVT(src, dst, R, KI, KP) \
    pad_cvt<<<((long)(R)*(KP) + 255) / 256, 256, 0, stream>>>((src), (dst), (R), (KI), (KP))
  CVT(Wih_f0, WihFB0,              1024, 300, 320);
  CVT(Wih_b0, WihFB0 + 1024L*320,  1024, 300, 320);
  CVT(Whh_f0, WhhFB0,              1024, 256, 256);
  CVT(Whh_b0, WhhFB0 + 1024L*256,  1024, 256, 256);
  CVT(Wih_f1, WihFB1,              1024, 256, 256);
  CVT(Wih_b1, WihFB1 + 1024L*256,  1024, 256, 256);
  CVT(Whh_f1, WhhFB1,              1024, 256, 256);
  CVT(Whh_b1, WhhFB1 + 1024L*256,  1024, 256, 256);
  CVT(W_caps, WcapsB,               256, 512, 512);
  #undef CVT

  // --- embedding gather ---
  {
    long total = 153600L * 320;
    embed_gather<<<(total + 255) / 256, 256, 0, stream>>>(tok, embedW, E_BF, total);
  }

  // --- word level ---
  // xp0 = e . Wih{f,b}^T : [153600,320] x [2048,320]^T (register-blocked 16x64/wave)
  gemm_bf16_x4<<<dim3(9600, 8), 128, 0, stream>>>(E_BF, WihFB0, XP0, 320, 2048);
  lstm_wmma<<<dim3(160, 2), 256, 0, stream>>>(XP0, WhhFB0, b_f0, b_b0, H1_BF, 2560, 60);
  // Wu1[n] = W_caps . h1[n]^T : batched [256,512] x [64,512]^T, store cols < 60
  gemm_bf16<<<dim3(16, 1, 2560), 128, 0, stream>>>(WcapsB, H1_BF, WU1,
                                                   512, 60, 60, 0, 64L * 512, 256L * 60);
  caps_route<<<2560, 256, 0, stream>>>(WU1, 60, CAP1F, CAP1B);

  // --- sentence level ---
  gemm_bf16_x4<<<dim3(160, 8), 128, 0, stream>>>(CAP1B, WihFB1, XP1, 256, 2048);
  lstm_wmma<<<dim3(8, 2), 256, 0, stream>>>(XP1, WhhFB1, b_f1, b_b1, H2_BF, 128, 20);
  gemm_bf16<<<dim3(16, 1, 128), 128, 0, stream>>>(WcapsB, H2_BF, WU2,
                                                  512, 20, 20, 0, 64L * 512, 256L * 20);
  caps_route<<<128, 256, 0, stream>>>(WU2, 20, CAP2F, CAP2B);

  // --- classifier ---
  fc_kernel<<<3, 256, 0, stream>>>(CAP2F, fcW, fcb, out);
}
